// ScaledDotProduct_71631464563512
// MI455X (gfx1250) — 8-bit matrix operand order repaired and checked by reading, compiled, not run on hardware
//
#include <hip/hip_runtime.h>

typedef __attribute__((ext_vector_type(16))) int   v16i;
typedef __attribute__((ext_vector_type(8)))  float v8f;

#define HEADS 16
#define SEQ   2048
#define DHEAD 128
#define KT_PER_WAVE 8
#define QSCALE 0.0859375f

__device__ __forceinline__ int pack4_fp8(float4 v) {
  int r = __builtin_amdgcn_cvt_pk_fp8_f32(v.x, v.y, 0, false);
  r     = __builtin_amdgcn_cvt_pk_fp8_f32(v.z, v.w, r, true);
  return r;
}

__device__ __forceinline__ float quant8(float x) {
  int p = __builtin_amdgcn_cvt_pk_fp8_f32(x, 0.0f, 0, false);
  return __builtin_amdgcn_cvt_f32_fp8(p, 0);
}

__global__ __launch_bounds__(256) void prep_fp8_kernel(const float* __restrict__ src,
                                                       unsigned int* __restrict__ dst, int n4) {
  int i = blockIdx.x * 256 + threadIdx.x;
  if (i < n4) dst[i] = (unsigned int)pack4_fp8(((const float4*)src)[i]);
}

__global__ __launch_bounds__(256) void prep_mask_kernel(const float* __restrict__ m,
                                                        float* __restrict__ qm, int n4) {
  int i = blockIdx.x * 256 + threadIdx.x;
  if (i < n4) {
    float4 v = ((const float4*)m)[i];
    int p = __builtin_amdgcn_cvt_pk_fp8_f32(v.x, v.y, 0, false);
    p     = __builtin_amdgcn_cvt_pk_fp8_f32(v.z, v.w, p, true);
    float4 o;
    o.x = __builtin_amdgcn_cvt_f32_fp8(p, 0);
    o.y = __builtin_amdgcn_cvt_f32_fp8(p, 1);
    o.z = __builtin_amdgcn_cvt_f32_fp8(p, 2);
    o.w = __builtin_amdgcn_cvt_f32_fp8(p, 3);
    ((float4*)qm)[i] = o;
  }
}

__global__ __launch_bounds__(256) void sdp_fp8_wmma_main(
    const unsigned int* __restrict__ q8,
    const unsigned int* __restrict__ k8,
    const float* __restrict__ qmask,
    const int*  __restrict__ kpm,
    float* __restrict__ out)
{
  const int lane = threadIdx.x & 31;
  const int wid  = threadIdx.x >> 5;
  const int unit = blockIdx.x * 8 + wid;

  const int kstrips = (SEQ / 16) / KT_PER_WAVE;
  const int kk = unit % kstrips;
  const int qt = (unit / kstrips) % (SEQ / 16);
  const int h  = unit / (kstrips * (SEQ / 16));

  const int half = lane >> 4;
  const int lrow = lane & 15;

  const unsigned int* qrp = q8 + (size_t)(h * SEQ + qt * 16 + lrow) * (DHEAD / 4);
  v16i a;
  #pragma unroll
  for (int g = 0; g < 2; ++g)
    #pragma unroll
    for (int p = 0; p < 4; ++p) {
      uint2 u = *(const uint2*)(qrp + 16 * g + 8 * (p >> 1) + 2 * (p & 1) + 4 * half);
      a[8 * g + 2 * p]     = (int)u.x;
      a[8 * g + 2 * p + 1] = (int)u.y;
    }

  const int ktile0 = kk * KT_PER_WAVE * 16;

  for (int t = 0; t < KT_PER_WAVE; ++t) {
    const int kt0 = ktile0 + t * 16;

    const unsigned int* krp = k8 + (size_t)(h * SEQ + kt0 + lrow) * (DHEAD / 4);
    v16i b;
    #pragma unroll
    for (int g = 0; g < 4; ++g) {
      uint4 u = *(const uint4*)(krp + 8 * g + 4 * half);
      b[4 * g]     = (int)u.x;
      b[4 * g + 1] = (int)u.y;
      b[4 * g + 2] = (int)u.z;
      b[4 * g + 3] = (int)u.w;
    }

    v8f c = {};
    c = __builtin_amdgcn_wmma_f32_16x16x128_fp8_fp8(a, b, (short)0, c, false, false);

    const int  ki   = kt0 + lrow;
    const bool kill = kpm[ki] != 0;
    const int  qi0  = qt * 16 + half * 8;
    const float* mrow = qmask + (size_t)qi0 * SEQ + ki;
    float*       orow = out + ((size_t)h * SEQ + qi0) * SEQ + ki;
    #pragma unroll
    for (int i = 0; i < 8; i += 2) {
      int p1 = __builtin_amdgcn_cvt_pk_fp8_f32(c[i], c[i + 1], 0, false);
      float s0 = __builtin_amdgcn_cvt_f32_fp8(p1, 0) * QSCALE;
      float s1 = __builtin_amdgcn_cvt_f32_fp8(p1, 1) * QSCALE;
      int p2 = __builtin_amdgcn_cvt_pk_fp8_f32(s0, s1, 0, false);
      s0 = __builtin_amdgcn_cvt_f32_fp8(p2, 0) + mrow[(size_t)i * SEQ];
      s1 = __builtin_amdgcn_cvt_f32_fp8(p2, 1) + mrow[(size_t)(i + 1) * SEQ];
      if (kill) { s0 = -__builtin_inff(); s1 = -__builtin_inff(); }
      orow[(size_t)i * SEQ]       = s0;
      orow[(size_t)(i + 1) * SEQ] = s1;
    }
  }
}

__global__ __launch_bounds__(256) void sdp_fp8_wmma_inline(
    const float* __restrict__ q,
    const float* __restrict__ kmat,
    const float* __restrict__ attn_mask,
    const int*  __restrict__ kpm,
    float* __restrict__ out)
{
  const int lane = threadIdx.x & 31;
  const int wid  = threadIdx.x >> 5;
  const int unit = blockIdx.x * 8 + wid;

  const int kstrips = (SEQ / 16) / KT_PER_WAVE;
  const int kk = unit % kstrips;
  const int qt = (unit / kstrips) % (SEQ / 16);
  const int h  = unit / (kstrips * (SEQ / 16));

  const int half = lane >> 4;
  const int lrow = lane & 15;

  const float* qrp = q + (((size_t)h * SEQ) + (size_t)(qt * 16 + lrow)) * DHEAD;
  v16i a;
  #pragma unroll
  for (int j = 0; j < 16; ++j) {
    const int jj = j & 7;
    const int kb = ((j >> 2) << 5) + (half << 4) + ((j & 3) << 2);
    a[j] = pack4_fp8(*(const float4*)(qrp + kb));
  }

  const int ktile0 = kk * KT_PER_WAVE * 16;
  for (int t = 0; t < KT_PER_WAVE; ++t) {
    const int kt0 = ktile0 + t * 16;
    const float* krp = kmat + (((size_t)h * SEQ) + (size_t)(kt0 + lrow)) * DHEAD;
    v16i b;
    #pragma unroll
    for (int j = 0; j < 16; ++j) {
      const int kb = ((j >> 2) << 5) + (half << 4) + ((j & 3) << 2);
      b[j] = pack4_fp8(*(const float4*)(krp + kb));
    }
    v8f c = {};
    c = __builtin_amdgcn_wmma_f32_16x16x128_fp8_fp8(a, b, (short)0, c, false, false);

    const int  ki   = kt0 + lrow;
    const bool kill = kpm[ki] != 0;
    #pragma unroll
    for (int i = 0; i < 8; i += 2) {
      const int qi = qt * 16 + (half << 3) + i;
      int p1 = __builtin_amdgcn_cvt_pk_fp8_f32(c[i], c[i + 1], 0, false);
      float s0 = __builtin_amdgcn_cvt_f32_fp8(p1, 0) * QSCALE;
      float s1 = __builtin_amdgcn_cvt_f32_fp8(p1, 1) * QSCALE;
      int p2 = __builtin_amdgcn_cvt_pk_fp8_f32(s0, s1, 0, false);
      s0 = __builtin_amdgcn_cvt_f32_fp8(p2, 0) + quant8(attn_mask[(size_t)qi * SEQ + ki]);
      s1 = __builtin_amdgcn_cvt_f32_fp8(p2, 1) + quant8(attn_mask[(size_t)(qi + 1) * SEQ + ki]);
      if (kill) { s0 = -__builtin_inff(); s1 = -__builtin_inff(); }
      out[(((size_t)h * SEQ) + qi) * SEQ + ki]     = s0;
      out[(((size_t)h * SEQ) + qi + 1) * SEQ + ki] = s1;
    }
  }
}

extern "C" void kernel_launch(void* const* d_in, const int* in_sizes, int n_in,
                              void* d_out, int out_size, void* d_ws, size_t ws_size,
                              hipStream_t stream) {
  const float* q         = (const float*)d_in[0];
  const float* k         = (const float*)d_in[1];
  const float* attn_mask = (const float*)d_in[2];
  const int*   kpm       = (const int*)d_in[3];
  float*       out       = (float*)d_out;

  const int units  = HEADS * (SEQ / 16) * ((SEQ / 16) / KT_PER_WAVE);
  const int blocks = units / 8;

  const size_t qk_bytes   = (size_t)HEADS * SEQ * DHEAD;
  const size_t mask_bytes = (size_t)SEQ * SEQ * sizeof(float);
  const size_t needed     = 2 * qk_bytes + mask_bytes;

  if (ws_size >= needed) {
    unsigned int* q8    = (unsigned int*)d_ws;
    unsigned int* k8    = q8 + qk_bytes / 4;
    float*        qmask = (float*)(k8 + qk_bytes / 4);

    const int n4qk = (int)(qk_bytes / 4);
    const int n4m  = SEQ * SEQ / 4;
    prep_fp8_kernel<<<(n4qk + 255) / 256, 256, 0, stream>>>(q, q8, n4qk);
    prep_fp8_kernel<<<(n4qk + 255) / 256, 256, 0, stream>>>(k, k8, n4qk);
    prep_mask_kernel<<<(n4m + 255) / 256, 256, 0, stream>>>(attn_mask, qmask, n4m);
    sdp_fp8_wmma_main<<<blocks, 256, 0, stream>>>(q8, k8, qmask, kpm, out);
  } else {
    sdp_fp8_wmma_inline<<<blocks, 256, 0, stream>>>(q, k, attn_mask, kpm, out);
  }
}
